// UniNetLoss_17944373362830
// MI455X (gfx1250) — matrix operand order repaired and checked by reading, not compiled, not run on hardware
//
#include <hip/hip_runtime.h>
#include <hip/hip_bf16.h>
#include <math.h>

typedef __bf16 bfv8  __attribute__((ext_vector_type(8)));   // 16 bytes
typedef __bf16 bfv16 __attribute__((ext_vector_type(16)));  // 32 bytes
typedef float  v8f   __attribute__((ext_vector_type(8)));

// ---------------------------------------------------------------------------
// Kernel 1: per-row stats. Row r = (b, i) of the [B*HW, C] logical matrix.
// Layout of src is NCHW, so element (r, c) lives at ((b*C + c)*HW + i):
// consecutive threads -> consecutive i -> fully coalesced loads per c.
// Outputs: 1/max(||s||,eps), 1/max(||t||,eps), and normalized dot (f32 exact).
// ---------------------------------------------------------------------------
__global__ void row_stats_kernel(const float* __restrict__ s,
                                 const float* __restrict__ t,
                                 int C, int HW,
                                 float* __restrict__ rns,
                                 float* __restrict__ rnt,
                                 float* __restrict__ dot) {
    int r = blockIdx.x * blockDim.x + threadIdx.x;   // level-local row
    int b = r / HW;
    int i = r % HW;
    const float* sb = s + (size_t)b * C * HW + i;
    const float* tb = t + (size_t)b * C * HW + i;
    float ss = 0.f, tt = 0.f, st = 0.f;
    for (int c = 0; c < C; ++c) {
        float a = sb[(size_t)c * HW];
        float d = tb[(size_t)c * HW];
        ss = fmaf(a, a, ss);
        tt = fmaf(d, d, tt);
        st = fmaf(a, d, st);
    }
    float rs = 1.f / fmaxf(sqrtf(ss), 1e-12f);
    float rt = 1.f / fmaxf(sqrtf(tt), 1e-12f);
    rns[r] = rs;
    rnt[r] = rt;
    dot[r] = st * rs * rt;   // sn . tn  (no 1/T here)
}

// ---------------------------------------------------------------------------
// Kernel 2: NCHW -> row-major [B*HW, C] bf16 with L2-normalization and the
// 1/sqrt(T) factor folded in. 32x32 LDS tile transpose for coalescing.
// Block (32, 8), each thread covers 4 tile rows.
// ---------------------------------------------------------------------------
__global__ void transpose_norm_kernel(const float* __restrict__ src,
                                      __bf16* __restrict__ dst,
                                      const float* __restrict__ rnorm,
                                      int C, int HW, float scale) {
    __shared__ float tile[32][33];
    int i0 = blockIdx.x * 32;
    int c0 = blockIdx.y * 32;
    int b  = blockIdx.z;
    int tx = threadIdx.x, ty = threadIdx.y;
#pragma unroll
    for (int yy = 0; yy < 4; ++yy) {
        int c = c0 + ty + yy * 8;
        tile[ty + yy * 8][tx] = src[((size_t)b * C + c) * HW + i0 + tx];
    }
    __syncthreads();
#pragma unroll
    for (int yy = 0; yy < 4; ++yy) {
        int irow = i0 + ty + yy * 8;
        float rn = rnorm[(size_t)b * HW + irow] * scale;
        dst[((size_t)b * HW + irow) * (size_t)C + c0 + tx] =
            (__bf16)(tile[tx][ty + yy * 8] * rn);
    }
}

// ---------------------------------------------------------------------------
// Kernel 3: fused GEMM + exp + row-sum.  rowsum[b,i] = sum_j exp(sn_i.tn_j/T)
// (the 1/T is pre-folded as 1/sqrt(T) into both sn and tn).
// One wave per 16-row tile; A fragments register-resident across the j sweep.
// 16-bit fragment k order used here (contiguous, same for A and B):
//   A (16xK): lane = (m = lane&15, h = lane>>4); VGPRs 0-7 hold
//             K=[h*16, h*16+16)  -> one 32B row chunk (two 16B loads).
//   B (Kx16): lane n = lane&15 selects column = tn row j0+n; VGPRs 0-7 hold
//             K=[h*16, h*16+16) -> one 32B row chunk.
//   C/D:      lane (n,h), VGPR v holds (M = h*8+v, N = n).
// ---------------------------------------------------------------------------
template <int KT>
__global__ void __launch_bounds__(128)
gemm_rowsum_kernel(const __bf16* __restrict__ sn,
                   const __bf16* __restrict__ tn,
                   float* __restrict__ rowsum, int HW) {
    constexpr int C = KT * 32;
    const int lane = threadIdx.x & 31;
    const int wave = threadIdx.x >> 5;
    const int tile = blockIdx.x * 4 + wave;          // 16-row tile index
    const int tilesPerBatch = HW >> 4;
    const int b  = tile / tilesPerBatch;
    const int i0 = (tile % tilesPerBatch) * 16;
    const int m  = lane & 15;
    const int h  = lane >> 4;

    const __bf16* Arow = sn + ((size_t)b * HW + i0 + m) * C;
    bfv16 afrag[KT];
#pragma unroll
    for (int kt = 0; kt < KT; ++kt) {
        union { bfv16 v; bfv8 p[2]; } u;
        u.p[0] = *(const bfv8*)(Arow + kt * 32 + h * 16);
        u.p[1] = *(const bfv8*)(Arow + kt * 32 + h * 16 + 8);
        afrag[kt] = u.v;
    }

    float rs[8];
#pragma unroll
    for (int v = 0; v < 8; ++v) rs[v] = 0.f;

    const __bf16* Bbase = tn + (size_t)b * HW * C + (size_t)m * C + h * 16;
    for (int j0 = 0; j0 < HW; j0 += 16) {
        v8f acc = {0.f, 0.f, 0.f, 0.f, 0.f, 0.f, 0.f, 0.f};
        const __bf16* bp = Bbase + (size_t)j0 * C;
#pragma unroll
        for (int kt = 0; kt < KT; ++kt) {
            bfv16 bfrag = *(const bfv16*)(bp + kt * 32);
            acc = __builtin_amdgcn_wmma_f32_16x16x32_bf16(
                false, afrag[kt], false, bfrag, (short)0, acc, false, false);
        }
#pragma unroll
        for (int v = 0; v < 8; ++v) rs[v] += __expf(acc[v]);
    }

    // Reduce each row's partials across the 16 lanes of its half-wave.
#pragma unroll
    for (int v = 0; v < 8; ++v) {
        float x = rs[v];
        x += __shfl_xor(x, 1, 32);
        x += __shfl_xor(x, 2, 32);
        x += __shfl_xor(x, 4, 32);
        x += __shfl_xor(x, 8, 32);
        rs[v] = x;
    }
    if ((lane & 15) == 0) {
        int base = b * HW + i0 + h * 8;
#pragma unroll
        for (int v = 0; v < 8; ++v) rowsum[base + v] = rs[v];
    }
}

// ---------------------------------------------------------------------------
// Kernel 4: final scalar reduction over all 21504 rows (3 levels).
// ---------------------------------------------------------------------------
__global__ void finalize_kernel(const float* __restrict__ dot,
                                const float* __restrict__ rowsum,
                                const int* __restrict__ marginp,
                                float* __restrict__ out) {
    const int rows[3] = {16384, 4096, 1024};
    const int base[3] = {0, 16384, 20480};
    float margin = (float)(*marginp);
    __shared__ float red[256];
    float total = 0.f;
    for (int l = 0; l < 3; ++l) {
        float cacc = 0.f, lacc = 0.f, macc = 0.f;
        for (int r = threadIdx.x; r < rows[l]; r += blockDim.x) {
            float d  = dot[base[l] + r];
            float dg = __expf(d * 0.5f) / (rowsum[base[l] + r] + 1e-8f);
            cacc += 1.f - d;
            lacc += -__logf(dg + 1e-8f);
            macc += fmaxf(margin - dg, 0.f);
        }
        float vals[3] = {cacc, lacc, macc};
        float res[3];
        for (int k = 0; k < 3; ++k) {
            red[threadIdx.x] = vals[k];
            __syncthreads();
            for (int s = 128; s > 0; s >>= 1) {
                if ((int)threadIdx.x < s) red[threadIdx.x] += red[threadIdx.x + s];
                __syncthreads();
            }
            res[k] = red[0];
            __syncthreads();
        }
        float inv = 1.f / (float)rows[l];
        total += 0.7f * res[0] * inv + 0.3f * res[1] * inv + res[2] * inv;
    }
    if (threadIdx.x == 0) out[0] = total;
}

// ---------------------------------------------------------------------------
extern "C" void kernel_launch(void* const* d_in, const int* in_sizes, int n_in,
                              void* d_out, int out_size, void* d_ws, size_t ws_size,
                              hipStream_t stream) {
    const float* s[3] = {(const float*)d_in[0], (const float*)d_in[1], (const float*)d_in[2]};
    const float* t[3] = {(const float*)d_in[3], (const float*)d_in[4], (const float*)d_in[5]};
    const int* margin = (const int*)d_in[6];

    const int B = 4;
    const int Cs[3]  = {256, 512, 512};
    const int HWs[3] = {4096, 1024, 256};
    const int rowsPer[3] = {B * 4096, B * 1024, B * 256};  // 16384, 4096, 1024
    const int basel[3] = {0, 16384, 20480};
    const int totalRows = 21504;

    char* ws = (char*)d_ws;
    size_t off = 0;
    void* snp[3];
    void* tnp[3];
    for (int l = 0; l < 3; ++l) {
        snp[l] = (void*)(ws + off); off += (size_t)B * HWs[l] * Cs[l] * 2;
        tnp[l] = (void*)(ws + off); off += (size_t)B * HWs[l] * Cs[l] * 2;
    }
    float* rns  = (float*)(ws + off); off += (size_t)totalRows * 4;
    float* rnt  = (float*)(ws + off); off += (size_t)totalRows * 4;
    float* dotb = (float*)(ws + off); off += (size_t)totalRows * 4;
    float* rsum = (float*)(ws + off); off += (size_t)totalRows * 4;

    const float scale = 0.70710678118f;  // 1/sqrt(T), T = 2.0 (folded into sn AND tn)

    for (int l = 0; l < 3; ++l) {
        int C = Cs[l], HW = HWs[l], rows = rowsPer[l];

        row_stats_kernel<<<rows / 256, 256, 0, stream>>>(
            s[l], t[l], C, HW, rns + basel[l], rnt + basel[l], dotb + basel[l]);

        dim3 tg(HW / 32, C / 32, B), tb(32, 8);
        transpose_norm_kernel<<<tg, tb, 0, stream>>>(
            s[l], (__bf16*)snp[l], rns + basel[l], C, HW, scale);
        transpose_norm_kernel<<<tg, tb, 0, stream>>>(
            t[l], (__bf16*)tnp[l], rnt + basel[l], C, HW, scale);

        int tiles = rows / 16;  // 1024 / 256 / 64 -> all divisible by 4 waves/block
        if (C == 256) {
            gemm_rowsum_kernel<8><<<tiles / 4, 128, 0, stream>>>(
                (const __bf16*)snp[l], (const __bf16*)tnp[l], rsum + basel[l], HW);
        } else {
            gemm_rowsum_kernel<16><<<tiles / 4, 128, 0, stream>>>(
                (const __bf16*)snp[l], (const __bf16*)tnp[l], rsum + basel[l], HW);
        }
    }

    finalize_kernel<<<1, 256, 0, stream>>>(dotb, rsum, margin, (float*)d_out);
}
